// ThreeLayerNet_1LevelAttn_54477365182710
// MI455X (gfx1250) — matrix operand order repaired and checked by reading, not compiled, not run on hardware
//
#include <hip/hip_runtime.h>
#include <hip/hip_bf16.h>

typedef __attribute__((ext_vector_type(16))) _Float16 v16h;
typedef __attribute__((ext_vector_type(8)))  _Float16 v8h;
typedef __attribute__((ext_vector_type(2)))  _Float16 v2h;
typedef __attribute__((ext_vector_type(8)))  float    v8f;

#define Bq  64
#define Uq  21
#define Vq  256
#define Dq  512
#define H1q 2048
#define H2q 512
#define Yq  10
#define XS  520   // padded LDS row stride in f16 (1040 B row pitch, de-banks row accesses)

// Build a 16x32 f16 A-fragment for v_wmma_f32_16x16x32_f16.
// rowptr must already point at  &A[row][k0 + (hiHalf ? 16 : 0)].
// Contiguous order: lanes 0-15 hold K 0..15, lanes 16-31 hold K 16..31.
__device__ __forceinline__ v16h load_a_frag(const _Float16* rowptr) {
  v8h lo = *(const v8h*)(rowptr);        // K j=0..7
  v8h hh = *(const v8h*)(rowptr + 8);    // K j=8..15 (at +8 elements)
  return __builtin_shufflevector(lo, hh, 0,1,2,3,4,5,6,7,8,9,10,11,12,13,14,15);
}

// Async-copy one 16 KB pre-swizzled B slab (global f16) into LDS.
// 256 threads x 16 B x 4 rounds. GVS addressing: saddr(SGPR pair) + vaddr(u32).
__device__ __forceinline__ void async_slab(const _Float16* gbase, unsigned ldsBase, int tid)
{
  unsigned long long g = (unsigned long long)gbase;
  unsigned voff = (unsigned)tid * 16u;
  #pragma unroll
  for (int rph = 0; rph < 4; ++rph) {
    asm volatile("global_load_async_to_lds_b128 %0, %1, %2"
                 :: "v"(ldsBase + voff + rph * 4096u),
                    "v"(voff + rph * 4096u),
                    "s"(g)
                 : "memory");
  }
}

// ---------------------------------------------------------------------------
// Kernel 0: pre-swizzle attn_w into fragment-ready f16 order (done once).
// AwSw[((ntile*16 + kt)*32 + lane)*16 + j] =
//      attn_w[ntile*16 + (lane&15)][kt*32 + (lane>=16?16:0) + j]
// ---------------------------------------------------------------------------
__global__ void swizzle_attnw(const float* __restrict__ Aw, _Float16* __restrict__ AwSw)
{
  int idx = blockIdx.x * blockDim.x + threadIdx.x;   // 0 .. 262143
  int j     = idx & 15;
  int lane  = (idx >> 4) & 31;
  int kt    = (idx >> 9) & 15;
  int ntile = idx >> 13;
  int n = ntile * 16 + (lane & 15);
  int k = kt * 32 + ((lane >> 4) << 4) + j;
  AwSw[idx] = (_Float16)Aw[(size_t)n * Dq + k];
}

// ---------------------------------------------------------------------------
// Kernel 1: fused bilinear attention per (b,u) block.
// 8 waves = 2 waves/SIMD (single block per WGP due to 295 KB LDS), so budget
// registers for exactly that occupancy: the wave's 32 frame-rows of A
// fragments (~256 VGPRs) stay register-resident; the inner loop touches LDS
// only for the B slab (1 ds_load_b128 per WMMA).
// ---------------------------------------------------------------------------
__global__ __launch_bounds__(256, 2)
void attn_kernel(const float* __restrict__ Xg,
                 const float* __restrict__ maskg,
                 const _Float16* __restrict__ AwSw,
                 _Float16* __restrict__ Xf)
{
  extern __shared__ __align__(32) char smem[];
  _Float16* Xh   = (_Float16*)smem;                          // Vq * XS f16
  _Float16* Bf   = (_Float16*)(smem + Vq * XS * 2);          // 2 x (16*32*16) f16 (double buffer)
  float*    Sarr = (float*)((char*)Bf + 2 * 16 * 32 * 16 * 2);
  float*    Warr = Sarr + Vq;
  float*    Red  = Warr + Vq;

  const int bu  = blockIdx.x;
  const int tid = threadIdx.x;
  const int wv  = tid >> 5;          // wave id 0..7
  const int l   = tid & 31;          // lane
  const int nl  = l & 15;
  const bool hihalf = l >= 16;

  const unsigned bfBase = (unsigned)(unsigned long long)Bf;  // LDS byte offset

  // Kick off async DMA of B slab 0 while we stage X.
  async_slab(AwSw, bfBase, tid);

  // ---- Stage X block into LDS as f16 (coalesced float4 reads) ----
  const float* Xblk = Xg + (size_t)bu * Vq * Dq;
  for (int r = 0; r < 32; ++r) {
    int v = wv * 32 + r;
    const float4* src = (const float4*)(Xblk + (size_t)v * Dq);
    #pragma unroll
    for (int c = 0; c < 4; ++c) {
      float4 f = src[l + c * 32];
      int col = (l + c * 32) * 4;
      _Float16* dst = Xh + v * XS + col;
      dst[0] = (_Float16)f.x; dst[1] = (_Float16)f.y;
      dst[2] = (_Float16)f.z; dst[3] = (_Float16)f.w;
    }
  }
  __syncthreads();

  // Two 16-row tiles per wave: rows [wv*16) and [(wv+8)*16)
  const int m0a = wv * 16;
  const int m0b = (wv + 8) * 16;
  const int kadd = hihalf ? 16 : 0;

  // ---- Load ALL A fragments for this wave's rows into registers ----
  v16h a0f[16], a1f[16];
  #pragma unroll
  for (int kt = 0; kt < 16; ++kt) {
    const int k0 = kt * 32 + kadd;
    a0f[kt] = load_a_frag(Xh + (m0a + nl) * XS + k0);
    a1f[kt] = load_a_frag(Xh + (m0b + nl) * XS + k0);
  }

  float s0[8], s1[8];
  #pragma unroll
  for (int r = 0; r < 8; ++r) { s0[r] = 0.f; s1[r] = 0.f; }

  for (int nt = 0; nt < 32; ++nt) {
    const int cur = nt & 1;
    const int n0  = nt * 16;
    if (nt + 1 < 32) {
      // Prefetch next slab into the other buffer, then wait for current slab
      // (async loads complete in order; 4 newest may stay outstanding).
      async_slab(AwSw + (size_t)(nt + 1) * 8192, bfBase + (unsigned)((cur ^ 1) * 16384), tid);
      asm volatile("s_wait_asynccnt 4" ::: "memory");
    } else {
      asm volatile("s_wait_asynccnt 0" ::: "memory");
    }
    __syncthreads();

    const _Float16* BfCur = Bf + cur * 8192;
    v8f c0 = {}; v8f c1 = {};
    #pragma unroll
    for (int kt = 0; kt < 16; ++kt) {
      v16h b = *(const v16h*)(BfCur + kt * 512 + l * 16);
      c0 = __builtin_amdgcn_wmma_f32_16x16x32_f16(false, a0f[kt], false, b, (short)0, c0, false, false);
      c1 = __builtin_amdgcn_wmma_f32_16x16x32_f16(false, a1f[kt], false, b, (short)0, c1, false, false);
    }

    // S_half tile (x) X tile, accumulate per-lane row partials
    #pragma unroll
    for (int r = 0; r < 8; ++r) {
      int rr = (hihalf ? 8 : 0) + r;
      s0[r] += c0[r] * (float)Xh[(m0a + rr) * XS + n0 + nl];
      s1[r] += c1[r] * (float)Xh[(m0b + rr) * XS + n0 + nl];
    }
    __syncthreads();
  }

  // Cross-lane column reduction (16 lanes per half) -> S per frame
  #pragma unroll
  for (int r = 0; r < 8; ++r) {
    float v0 = s0[r], v1 = s1[r];
    for (int off = 1; off < 16; off <<= 1) {
      v0 += __shfl_xor(v0, off, 16);
      v1 += __shfl_xor(v1, off, 16);
    }
    if (nl == 0) {
      int rr = (hihalf ? 8 : 0) + r;
      Sarr[m0a + rr] = v0;
      Sarr[m0b + rr] = v1;
    }
  }
  __syncthreads();

  // Softmax over frames of (tanh(S)+1)*mask
  float st = (tanhf(Sarr[tid]) + 1.0f) * maskg[(size_t)bu * Vq + tid];
  Red[tid] = st; __syncthreads();
  for (int s = 128; s > 0; s >>= 1) { if (tid < s) Red[tid] = fmaxf(Red[tid], Red[tid + s]); __syncthreads(); }
  float mx = Red[0]; __syncthreads();
  float e = __expf(st - mx);
  Red[tid] = e; __syncthreads();
  for (int s = 128; s > 0; s >>= 1) { if (tid < s) Red[tid] += Red[tid + s]; __syncthreads(); }
  float inv = 1.0f / Red[0];
  Warr[tid] = e * inv;
  __syncthreads();

  // Weighted sum over frames -> x_attn (f16) into Xf[b][u*D + d]
  // Thread tid owns the f16 pair (d, d+1) with d = 2*tid: one b32 LDS load/frame.
  const int b_idx = bu / Uq, u_idx = bu % Uq;
  _Float16* outp = Xf + (size_t)b_idx * (Uq * Dq) + (size_t)u_idx * Dq;
  {
    const int d = 2 * tid;
    float acc0 = 0.f, acc1 = 0.f;
    for (int v = 0; v < Vq; ++v) {
      v2h p = *(const v2h*)(Xh + v * XS + d);
      float w = Warr[v];
      acc0 += w * (float)p.x;
      acc1 += w * (float)p.y;
    }
    outp[d]     = (_Float16)acc0;
    outp[d + 1] = (_Float16)acc1;
  }
}

// ---------------------------------------------------------------------------
// Kernel 2/3: out = relu(A @ W^T + bias), M = 64 fixed.
// One block per 16-col tile; 4 waves split K; each wave does all 4 m-tiles
// (so W is streamed exactly once and each B fragment feeds 4 WMMAs).
// ---------------------------------------------------------------------------
__global__ __launch_bounds__(128)
void gemm_wmma(const _Float16* __restrict__ A,
               const float* __restrict__ W,
               const float* __restrict__ bias,
               _Float16* __restrict__ out,
               int N, int K, int do_relu)
{
  __shared__ float Red[4][4][32][8];   // [kslice][mtile][lane][r] = 16 KB
  const int n0  = blockIdx.x * 16;
  const int tid = threadIdx.x;
  const int wv  = tid >> 5;            // K-slice 0..3
  const int l   = tid & 31;
  const int nl  = l & 15;
  const bool hihalf = l >= 16;
  const int kaddA = hihalf ? 16 : 0;
  const int kaddB = hihalf ? 16 : 0;
  const int kq   = K >> 2;
  const int kbeg = wv * kq;
  const int kend = kbeg + kq;

  const float* wrow = W + (size_t)(n0 + nl) * K;

  v8f c[4] = {};
  for (int k0 = kbeg; k0 < kend; k0 += 32) {
    if (k0 + 64 < kend) __builtin_prefetch(wrow + k0 + 64, 0, 1);  // global_prefetch_b8
    const float* wp = wrow + k0 + kaddB;
    v16h b;
    #pragma unroll
    for (int j = 0; j < 16; ++j) b[j] = (_Float16)wp[j];
    #pragma unroll
    for (int m = 0; m < 4; ++m) {
      v16h a = load_a_frag(A + (size_t)(m * 16 + nl) * K + k0 + kaddA);
      c[m] = __builtin_amdgcn_wmma_f32_16x16x32_f16(false, a, false, b, (short)0, c[m], false, false);
    }
  }

  #pragma unroll
  for (int m = 0; m < 4; ++m)
    #pragma unroll
    for (int r = 0; r < 8; ++r) Red[wv][m][l][r] = c[m][r];
  __syncthreads();

  // Wave wv finalizes m-tile wv.
  {
    const int m = wv;
    float bb = bias[n0 + nl];
    #pragma unroll
    for (int r = 0; r < 8; ++r) {
      float v = Red[0][m][l][r] + Red[1][m][l][r] + Red[2][m][l][r] + Red[3][m][l][r] + bb;
      if (do_relu) v = fmaxf(v, 0.f);
      int row = m * 16 + (hihalf ? 8 : 0) + r;
      out[(size_t)row * N + n0 + nl] = (_Float16)v;
    }
  }
}

// ---------------------------------------------------------------------------
// Kernel 4: tiny final layer (N=10 < 16) -> plain VALU, fp32 output.
// ---------------------------------------------------------------------------
__global__ void final_fc(const _Float16* __restrict__ h2,
                         const float* __restrict__ W3,
                         const float* __restrict__ b3,
                         float* __restrict__ out)
{
  int idx = blockIdx.x * blockDim.x + threadIdx.x;
  if (idx >= Bq * Yq) return;
  int bi = idx / Yq, y = idx % Yq;
  const _Float16* hp = h2 + (size_t)bi * H2q;
  const float*    wp = W3 + (size_t)y * H2q;
  float acc = b3[y];
  for (int k = 0; k < H2q; ++k) acc += (float)hp[k] * wp[k];
  out[idx] = acc;
}

extern "C" void kernel_launch(void* const* d_in, const int* in_sizes, int n_in,
                              void* d_out, int out_size, void* d_ws, size_t ws_size,
                              hipStream_t stream)
{
  const float* X     = (const float*)d_in[0];
  const float* mask  = (const float*)d_in[1];
  const float* attnw = (const float*)d_in[2];
  const float* W1    = (const float*)d_in[3];
  const float* b1    = (const float*)d_in[4];
  const float* W2    = (const float*)d_in[5];
  const float* b2    = (const float*)d_in[6];
  const float* W3    = (const float*)d_in[7];
  const float* b3    = (const float*)d_in[8];
  float* out = (float*)d_out;

  _Float16* Xf   = (_Float16*)d_ws;                     // [64][10752] f16
  _Float16* h1   = Xf + (size_t)Bq * Uq * Dq;           // [64][2048]  f16
  _Float16* h2   = h1 + (size_t)Bq * H1q;               // [64][512]   f16
  _Float16* AwSw = h2 + (size_t)Bq * H2q;               // [512*512]   f16 pre-swizzled

  swizzle_attnw<<<dim3((Dq * Dq) / 256), dim3(256), 0, stream>>>(attnw, AwSw);

  size_t smem = (size_t)Vq * XS * 2          // X block f16
              + (size_t)2 * 16 * 32 * 16 * 2 // double-buffered B fragment slab
              + (size_t)3 * Vq * 4;          // S, W, reduction
  attn_kernel<<<dim3(Bq * Uq), dim3(256), smem, stream>>>(X, mask, AwSw, Xf);

  gemm_wmma<<<dim3(H1q / 16), dim3(128), 0, stream>>>(Xf, W1, b1, h1, H1q, Uq * Dq, 1);
  gemm_wmma<<<dim3(H2q / 16), dim3(128), 0, stream>>>(h1, W2, b2, h2, H2q, H1q, 1);
  final_fc<<<dim3((Bq * Yq + 127) / 128), dim3(128), 0, stream>>>(h2, W3, b3, out);
}
